// ApplyKmeans_55989193670839
// MI455X (gfx1250) — matrix operand order repaired and checked by reading, not compiled, not run on hardware
//
#include <hip/hip_runtime.h>

// ---------------------------------------------------------------------------
// K-means assignment for MI455X (gfx1250, wave32, WMMA).
// tokens[n] = argmin_k ||x_n - c_k||^2  ==  argmax_k ( x_n . c_k - 0.5*Cnorm[k] )
// GEMM done with split-bf16 (hi+lo) v_wmma_f32_16x16x32_bf16, 3 independent
// accumulator chains, double-buffered A(LDS)/B(L2) fragment loads.
// ---------------------------------------------------------------------------

#define N_ROWS   131072
#define DIMD     1024
#define KC       300
#define KTILES   19            // ceil(300/16)
#define DCHUNKS  32            // 1024 / 32
#define NFRAG    (KTILES * DCHUNKS)   // 608 fragments per plane
#define FRAG_BYTES 1024        // 32 lanes * 32 B
#define ROWS_WG  32
#define RPAD     1032          // 1024 + 8 ushort pad -> kills LDS bank conflicts

typedef __attribute__((ext_vector_type(16))) __bf16 v16bf;
typedef __attribute__((ext_vector_type(8)))  float  v8f;

union FragU { uint4 u[2]; v16bf v; };

__device__ __forceinline__ unsigned short f32_to_bf16_rne(float f) {
  unsigned u = __float_as_uint(f);
  u += 0x7fffu + ((u >> 16) & 1u);
  return (unsigned short)(u >> 16);
}
__device__ __forceinline__ float bf16_bits_to_f32(unsigned short h) {
  return __uint_as_float(((unsigned)h) << 16);
}

// ---------------------------------------------------------------------------
// Prep 1: pack C [D=1024, K=300] fp32 into bf16 hi/lo WMMA B-fragments.
// Fragment fi = kt*32 + dc covers cols [kt*16, kt*16+16), K-depth [dc*32, dc*32+32).
// B layout (32x16 bf16): lane holds col = lane&15, K = kb+j (j<8) / 16+kb+j-8 (j>=8), kb = (lane<16)?0:8.
// Each lane's 16 bf16 stored contiguously (32 B) -> coalesced fragment loads.
// ---------------------------------------------------------------------------
__global__ void prep_cpack(const float* __restrict__ C,
                           unsigned short* __restrict__ cp_hi,
                           unsigned short* __restrict__ cp_lo) {
  const int fi   = blockIdx.x;          // 0..607
  const int lane = threadIdx.x;         // 0..31
  const int kt = fi >> 5, dc = fi & 31;
  const int col = lane & 15;
  const int k   = kt * 16 + col;
  const int kb  = (lane < 16) ? 0 : 8;

  alignas(16) unsigned short th[16], tl[16];
#pragma unroll
  for (int j = 0; j < 16; ++j) {
    const int d = dc * 32 + kb + (j < 8 ? j : j + 8);
    const float v = (k < KC) ? C[(size_t)d * KC + k] : 0.0f;
    const unsigned short hb = f32_to_bf16_rne(v);
    th[j] = hb;
    tl[j] = f32_to_bf16_rne(v - bf16_bits_to_f32(hb));
  }
  const size_t off = ((size_t)fi * 32 + lane) * 16;   // ushort units (32 B / lane)
  ((uint4*)(cp_hi + off))[0] = ((const uint4*)th)[0];
  ((uint4*)(cp_hi + off))[1] = ((const uint4*)th)[1];
  ((uint4*)(cp_lo + off))[0] = ((const uint4*)tl)[0];
  ((uint4*)(cp_lo + off))[1] = ((const uint4*)tl)[1];
}

// Prep 2: bias[k] = -0.5*Cnorm[k], -inf for padded columns (never win argmax).
__global__ void prep_bias(const float* __restrict__ Cnorm, float* __restrict__ bias) {
  const int k = blockIdx.x * blockDim.x + threadIdx.x;
  if (k < KTILES * 16)
    bias[k] = (k < KC) ? (-0.5f * Cnorm[k]) : -__builtin_inff();
}

// ---------------------------------------------------------------------------
// Main: 256 threads = 8 waves per WG, WG owns 32 rows (full D in LDS as bf16
// hi/lo).  wave w: row-tile rt = w&1, kt-group g = w>>1 covers ~5 column tiles.
// ---------------------------------------------------------------------------
__global__ void __launch_bounds__(256)
kmeans_assign(const float* __restrict__ x,
              const unsigned short* __restrict__ cp_hi,
              const unsigned short* __restrict__ cp_lo,
              const float* __restrict__ bias,
              int* __restrict__ tokens) {
  extern __shared__ unsigned short smem[];
  unsigned short* sh_hi = smem;                       // 32 x 1032 ushort
  unsigned short* sh_lo = smem + ROWS_WG * RPAD;      // 32 x 1032 ushort
  float* redv = (float*)(smem + 2 * ROWS_WG * RPAD);  // [4][32]
  int*   redk = (int*)(redv + 4 * ROWS_WG);           // [4][32]

  const int tid = threadIdx.x;
  const int rowbase = blockIdx.x * ROWS_WG;

  // --- load x slab (32 rows x 1024 fp32), split to bf16 hi/lo in LDS ---
  for (int it = 0; it < ROWS_WG; ++it) {
    const float4 xv = *(const float4*)(x + (size_t)(rowbase + it) * DIMD + tid * 4);
    const float f[4] = {xv.x, xv.y, xv.z, xv.w};
    const int base = it * RPAD + tid * 4;
#pragma unroll
    for (int j = 0; j < 4; ++j) {
      const unsigned short hb = f32_to_bf16_rne(f[j]);
      sh_hi[base + j] = hb;
      sh_lo[base + j] = f32_to_bf16_rne(f[j] - bf16_bits_to_f32(hb));
    }
  }
  __syncthreads();

  const int lane = tid & 31;
  const int w    = tid >> 5;
  const int rt   = w & 1;        // row tile within WG
  const int g    = w >> 1;       // kt group 0..3
  const int l15  = lane & 15;
  const int hiHalf = lane >> 4;
  const int loA  = hiHalf ? 8 : 0;            // A-frag K-sub-offset for this lane
  const int abase = (rt * 16 + l15) * RPAD;   // A row base in LDS

  const int kt0 = g * 5;
  const int kt1 = (kt0 + 5 < KTILES) ? (kt0 + 5) : KTILES;

  float best[8];
  int   bcol[8];
#pragma unroll
  for (int v = 0; v < 8; ++v) { best[v] = -__builtin_inff(); bcol[v] = 0; }

  for (int kt = kt0; kt < kt1; ++kt) {
    v8f hh = {}, hl = {}, lh = {};
    const char* bh_base = (const char*)cp_hi + (size_t)kt * DCHUNKS * FRAG_BYTES + (size_t)lane * 32;
    const char* bl_base = (const char*)cp_lo + (size_t)kt * DCHUNKS * FRAG_BYTES + (size_t)lane * 32;

    // --- double-buffered fragment pipeline: issue loads for dc+1 before
    //     consuming dc, so WMMA never drains loadcnt/dscnt to zero ---
    FragU a_h[2], a_l[2], b_h[2], b_l[2];
    {
      const int ao = abase + loA;
      a_h[0].u[0] = *(const uint4*)(sh_hi + ao);
      a_h[0].u[1] = *(const uint4*)(sh_hi + ao + 16);
      a_l[0].u[0] = *(const uint4*)(sh_lo + ao);
      a_l[0].u[1] = *(const uint4*)(sh_lo + ao + 16);
      const uint4* ph = (const uint4*)bh_base;
      const uint4* pl = (const uint4*)bl_base;
      b_h[0].u[0] = ph[0]; b_h[0].u[1] = ph[1];
      b_l[0].u[0] = pl[0]; b_l[0].u[1] = pl[1];
    }

#pragma unroll 4
    for (int dc = 0; dc < DCHUNKS; ++dc) {
      const int cur = dc & 1;
      const int nxt = cur ^ 1;
      if (dc + 1 < DCHUNKS) {
        const int ao = abase + (dc + 1) * 32 + loA;
        a_h[nxt].u[0] = *(const uint4*)(sh_hi + ao);
        a_h[nxt].u[1] = *(const uint4*)(sh_hi + ao + 16);
        a_l[nxt].u[0] = *(const uint4*)(sh_lo + ao);
        a_l[nxt].u[1] = *(const uint4*)(sh_lo + ao + 16);
        const uint4* ph = (const uint4*)(bh_base + (dc + 1) * FRAG_BYTES);
        const uint4* pl = (const uint4*)(bl_base + (dc + 1) * FRAG_BYTES);
        b_h[nxt].u[0] = ph[0]; b_h[nxt].u[1] = ph[1];
        b_l[nxt].u[0] = pl[0]; b_l[nxt].u[1] = pl[1];
      }
      // three independent accumulator chains: xh*ch + xh*cl + xl*ch
      hh = __builtin_amdgcn_wmma_f32_16x16x32_bf16(false, a_h[cur].v, false, b_h[cur].v,
                                                   (short)0, hh, false, false);
      hl = __builtin_amdgcn_wmma_f32_16x16x32_bf16(false, a_h[cur].v, false, b_l[cur].v,
                                                   (short)0, hl, false, false);
      lh = __builtin_amdgcn_wmma_f32_16x16x32_bf16(false, a_l[cur].v, false, b_h[cur].v,
                                                   (short)0, lh, false, false);
    }

    const v8f s = hh + hl + lh;
    const float bi = bias[kt * 16 + l15];
    const int col = kt * 16 + l15;
#pragma unroll
    for (int v = 0; v < 8; ++v) {
      const float val = s[v] + bi;   // score = x.c - 0.5*||c||^2  (argmax)
      if (val > best[v]) { best[v] = val; bcol[v] = col; }
    }
  }

  // cross-lane argmax within each 16-lane half (C/D layout: lane = col, VGPR = row)
#pragma unroll
  for (int v = 0; v < 8; ++v) {
    float bv = best[v];
    int   bk = bcol[v];
#pragma unroll
    for (int m = 1; m < 16; m <<= 1) {
      const float ov = __shfl_xor(bv, m, 32);
      const int   ok = __shfl_xor(bk, m, 32);
      if (ov > bv || (ov == bv && ok < bk)) { bv = ov; bk = ok; }
    }
    if (l15 == 0) {
      const int rl = rt * 16 + hiHalf * 8 + v;   // local row 0..31
      redv[g * ROWS_WG + rl] = bv;
      redk[g * ROWS_WG + rl] = bk;
    }
  }
  __syncthreads();

  // merge the 4 kt-groups, first-min tie-break, emit int32 token
  if (tid < ROWS_WG) {
    float bv = redv[tid];
    int   bk = redk[tid];
#pragma unroll
    for (int gg = 1; gg < 4; ++gg) {
      const float ov = redv[gg * ROWS_WG + tid];
      const int   ok = redk[gg * ROWS_WG + tid];
      if (ov > bv || (ov == bv && ok < bk)) { bv = ov; bk = ok; }
    }
    tokens[rowbase + tid] = bk;
  }
}

// ---------------------------------------------------------------------------
extern "C" void kernel_launch(void* const* d_in, const int* in_sizes, int n_in,
                              void* d_out, int out_size, void* d_ws, size_t ws_size,
                              hipStream_t stream) {
  (void)in_sizes; (void)n_in; (void)out_size; (void)ws_size;
  const float* x     = (const float*)d_in[0];   // [131072,1024] f32
  const float* C     = (const float*)d_in[1];   // [1024,300]   f32
  const float* Cnorm = (const float*)d_in[2];   // [300]        f32
  int* tokens = (int*)d_out;                    // [32,4096] int32 (flat 131072)

  // workspace layout
  const size_t plane_ush = (size_t)NFRAG * FRAG_BYTES / 2;  // ushort count per plane
  unsigned short* cp_hi = (unsigned short*)d_ws;
  unsigned short* cp_lo = cp_hi + plane_ush;
  float* bias = (float*)(cp_lo + plane_ush);                // 19*16 floats

  const int SHBYTES = 2 * ROWS_WG * RPAD * (int)sizeof(unsigned short)
                    + 4 * ROWS_WG * (int)(sizeof(float) + sizeof(int));  // 133120
  hipFuncSetAttribute(reinterpret_cast<const void*>(kmeans_assign),
                      hipFuncAttributeMaxDynamicSharedMemorySize, SHBYTES);

  prep_cpack<<<NFRAG, 32, 0, stream>>>(C, cp_hi, cp_lo);
  prep_bias<<<1, 320, 0, stream>>>(Cnorm, bias);
  kmeans_assign<<<N_ROWS / ROWS_WG, 256, SHBYTES, stream>>>(x, cp_hi, cp_lo, bias, tokens);
}
